// KNNAugmentedGPT2_90211493085599
// MI455X (gfx1250) — matrix operand order repaired and checked by reading, not compiled, not run on hardware
//
#include <hip/hip_runtime.h>
#include <stdint.h>

// ---------------- problem constants (match reference) ----------------
#define B_    64
#define D_    768
#define N_    262144
#define V_    50257
#define KNN   8
#define TEMPK 10.0f
#define LAM   0.25f
#define EPSK  1e-10f

// ---------------- tiling ----------------
#define NWG    128          // workgroups in GEMM kernel
#define CHUNK  2048         // keys per workgroup
#define TILE_N 128          // keys per tile
#define TILES  (CHUNK / TILE_N)
#define KSTEPS (D_ / 32)    // 24 k-steps of 32 (bf16 WMMA K)
#define DT_P   133          // padded pitch of distance tile (bank-conflict break)

typedef __bf16 bf16_t;
typedef bf16_t v16bf __attribute__((ext_vector_type(16)));
typedef float  v8f   __attribute__((ext_vector_type(8)));

union OpPack {            // 16 bf16 per lane = two 16B chunks
    v16bf  v;
    float4 f4[2];
};

static __device__ __forceinline__ unsigned short f2bf(float x) {
    unsigned u = __float_as_uint(x);
    u += 0x7FFFu + ((u >> 16) & 1u);          // round-to-nearest-even
    return (unsigned short)(u >> 16);
}
static __device__ __forceinline__ unsigned pack_bf2(float a, float b) {
    return (unsigned)f2bf(a) | ((unsigned)f2bf(b) << 16);
}

// ============ kernel 0: queries -> bf16 scratch + ||q||^2 ============
__global__ void __launch_bounds__(256)
convert_q(const float* __restrict__ q, unsigned short* __restrict__ qbf,
          float* __restrict__ q2g) {
    __shared__ float red[256];
    const int b = blockIdx.x, t = threadIdx.x;
    float s = 0.f;
#pragma unroll
    for (int i = 0; i < D_ / 256; ++i) {
        int c = t + i * 256;
        float x = q[(size_t)b * D_ + c];
        qbf[(size_t)b * D_ + c] = f2bf(x);
        s += x * x;
    }
    red[t] = s; __syncthreads();
    for (int w = 128; w > 0; w >>= 1) {
        if (t < w) red[t] += red[t + w];
        __syncthreads();
    }
    if (t == 0) q2g[b] = red[0];
}

// ============ kernel 1: streamed bf16 WMMA GEMM + fused top-8 ============
static __device__ __forceinline__ v8f wmma_acc(const unsigned short* ap,
                                               const v16bf& bv, v8f c) {
    OpPack a;
    a.f4[0] = *(const float4*)(ap);        // K = kb .. kb+7
    a.f4[1] = *(const float4*)(ap + 16);   // K = kb+16 .. kb+23
    return __builtin_amdgcn_wmma_f32_16x16x32_bf16(
        false, a.v, false, bv, (short)0, c, false, false);
}

static __device__ __forceinline__ void store_dist(float* dt, const v8f& a, int mt,
                                                  int ncol, int mhi,
                                                  const float* q2s, float k2v) {
#pragma unroll
    for (int j = 0; j < 8; ++j) {
        int m = mt * 16 + j + mhi;                 // C layout: M = j + 8*(lane>>4)
        dt[m * DT_P + ncol] = q2s[m] + k2v - 2.0f * a[j];
    }
}

__global__ void __launch_bounds__(256)
knn_gemm_topk(const float* __restrict__ keys, const unsigned short* __restrict__ qbf,
              const float* __restrict__ q2g,
              float* __restrict__ cand_d, int* __restrict__ cand_i) {
    __shared__ __align__(16) unsigned short ks[TILE_N * 40]; // 128 keys x 32 bf16 (+pad)
    __shared__ float pk2[256];
    __shared__ float k2s[TILE_N];
    __shared__ float q2s[B_];
    __shared__ __align__(16) float dt[64 * DT_P];            // also merge buffer
    float* mbd = dt;                                          // [64][4][8] float
    int*   mbi = (int*)(dt + 2048);                           // [64][4][8] int

    const int t = threadIdx.x, lane = t & 31, w = t >> 5;
    const int n_base = blockIdx.x * CHUNK;
    if (t < B_) q2s[t] = q2g[t];

    // private running top-8 (fully unrolled -> registers)
    float bd[KNN]; int bi[KNN];
#pragma unroll
    for (int j = 0; j < KNN; ++j) { bd[j] = 3.0e38f; bi[j] = 0; }
    float bmax = 3.0e38f; int bmp = 0;

    const int srow = t >> 1, shalf = t & 1;     // key-block loading role
    const int m_scan = t & 63, part = t >> 6;   // scan role

    for (int tile = 0; tile < TILES; ++tile) {
        const int n0 = n_base + tile * TILE_N;
        __syncthreads();                         // prev tile fully consumed
        if (t < TILE_N) k2s[t] = 0.f;

        v8f acc0 = {}, acc1 = {}, acc2 = {}, acc3 = {};

        for (int kk = 0; kk < KSTEPS; ++kk) {
            const int K0 = kk * 32;
            // ---- load 128x32 f32 keys, convert to bf16 into LDS + |k|^2 partials
            {
                const float* gp = keys + (size_t)(n0 + srow) * D_ + K0 + shalf * 16;
                float4 x0 = ((const float4*)gp)[0];
                float4 x1 = ((const float4*)gp)[1];
                float4 x2 = ((const float4*)gp)[2];
                float4 x3 = ((const float4*)gp)[3];
                if (kk + 1 < KSTEPS) __builtin_prefetch(gp + 32, 0, 1);
                float ssq = x0.x*x0.x + x0.y*x0.y + x0.z*x0.z + x0.w*x0.w
                          + x1.x*x1.x + x1.y*x1.y + x1.z*x1.z + x1.w*x1.w
                          + x2.x*x2.x + x2.y*x2.y + x2.z*x2.z + x2.w*x2.w
                          + x3.x*x3.x + x3.y*x3.y + x3.z*x3.z + x3.w*x3.w;
                unsigned* kp = (unsigned*)(ks + srow * 40 + shalf * 16);
                kp[0] = pack_bf2(x0.x, x0.y); kp[1] = pack_bf2(x0.z, x0.w);
                kp[2] = pack_bf2(x1.x, x1.y); kp[3] = pack_bf2(x1.z, x1.w);
                kp[4] = pack_bf2(x2.x, x2.y); kp[5] = pack_bf2(x2.z, x2.w);
                kp[6] = pack_bf2(x3.x, x3.y); kp[7] = pack_bf2(x3.z, x3.w);
                pk2[t] = ssq;
            }
            __syncthreads();
            if (t < TILE_N) k2s[t] += pk2[2 * t] + pk2[2 * t + 1];
            // ---- WMMA: wave w owns 16-key B tile; loops 4 query row-tiles
            {
                OpPack b;
                const unsigned short* bp =
                    ks + (w * 16 + (lane & 15)) * 40 + ((lane >> 4) * 8);
                b.f4[0] = *(const float4*)bp;        // K = kr .. kr+7
                b.f4[1] = *(const float4*)(bp + 16); // K = kr+16 .. kr+23
                const int arow = lane & 15;
                const int kb = K0 + ((lane >> 4) * 8);
                const unsigned short* a0 = qbf + (size_t)(0 * 16 + arow) * D_ + kb;
                const unsigned short* a1 = qbf + (size_t)(1 * 16 + arow) * D_ + kb;
                const unsigned short* a2 = qbf + (size_t)(2 * 16 + arow) * D_ + kb;
                const unsigned short* a3 = qbf + (size_t)(3 * 16 + arow) * D_ + kb;
                acc0 = wmma_acc(a0, b.v, acc0);
                acc1 = wmma_acc(a1, b.v, acc1);
                acc2 = wmma_acc(a2, b.v, acc2);
                acc3 = wmma_acc(a3, b.v, acc3);
            }
            __syncthreads();                     // protect ks for next k-step
        }
        // ---- distances -> LDS tile
        {
            const int ncol = w * 16 + (lane & 15);
            const int mhi = 8 * (lane >> 4);
            const float k2v = k2s[ncol];
            store_dist(dt, acc0, 0, ncol, mhi, q2s, k2v);
            store_dist(dt, acc1, 1, ncol, mhi, q2s, k2v);
            store_dist(dt, acc2, 2, ncol, mhi, q2s, k2v);
            store_dist(dt, acc3, 3, ncol, mhi, q2s, k2v);
        }
        __syncthreads();
        // ---- each thread scans 32 columns of its row, keeps running top-8
        {
            const float* row = dt + m_scan * DT_P + part * 32;
            const int idx0 = n0 + part * 32;
            for (int j = 0; j < 32; ++j) {
                float d = row[j];
                if (d < bmax) {
                    int id = idx0 + j;
#pragma unroll
                    for (int qq = 0; qq < KNN; ++qq)
                        if (qq == bmp) { bd[qq] = d; bi[qq] = id; }
                    bmax = bd[0]; bmp = 0;
#pragma unroll
                    for (int qq = 1; qq < KNN; ++qq)
                        if (bd[qq] > bmax) { bmax = bd[qq]; bmp = qq; }
                }
            }
        }
    }
    __syncthreads();
    // ---- in-WG merge: 4 partial top-8s per row -> top-8, write candidates
    {
        float* pd = mbd + (m_scan * 4 + part) * 8;
        int*   pi = mbi + (m_scan * 4 + part) * 8;
#pragma unroll
        for (int j = 0; j < KNN; ++j) { pd[j] = bd[j]; pi[j] = bi[j]; }
    }
    __syncthreads();
    if (t < B_) {
        float fd[KNN]; int fi[KNN];
#pragma unroll
        for (int j = 0; j < KNN; ++j) { fd[j] = 3.0e38f; fi[j] = 0; }
        float fmax = 3.0e38f; int fmp = 0;
        const float* pd = mbd + t * 32;
        const int*   pi = mbi + t * 32;
        for (int g = 0; g < 32; ++g) {
            float d = pd[g];
            if (d < fmax) {
                int id = pi[g];
#pragma unroll
                for (int qq = 0; qq < KNN; ++qq)
                    if (qq == fmp) { fd[qq] = d; fi[qq] = id; }
                fmax = fd[0]; fmp = 0;
#pragma unroll
                for (int qq = 1; qq < KNN; ++qq)
                    if (fd[qq] > fmax) { fmax = fd[qq]; fmp = qq; }
            }
        }
        float* od = cand_d + ((size_t)t * NWG + blockIdx.x) * KNN;
        int*   oi = cand_i + ((size_t)t * NWG + blockIdx.x) * KNN;
#pragma unroll
        for (int j = 0; j < KNN; ++j) { od[j] = fd[j]; oi[j] = fi[j]; }
    }
}

// ============ kernel 2: merge candidates, softmaxes, outputs ============
__global__ void __launch_bounds__(256)
knn_finalize(const float* __restrict__ base, const int* __restrict__ vals,
             const float* __restrict__ cand_d, const int* __restrict__ cand_i,
             float* __restrict__ out) {
    __shared__ float cd[NWG * KNN];
    __shared__ int   ci[NWG * KNN];
    __shared__ float red[256];
    __shared__ float stok_w[KNN];
    __shared__ int   stok_id[KNN];

    const int b = blockIdx.x, t = threadIdx.x;
    for (int i = t; i < NWG * KNN; i += 256) {
        cd[i] = cand_d[(size_t)b * NWG * KNN + i];
        ci[i] = cand_i[(size_t)b * NWG * KNN + i];
    }
    __syncthreads();
    if (t == 0) {
        float d8[KNN]; int i8[KNN];
#pragma unroll
        for (int j = 0; j < KNN; ++j) { d8[j] = 3.0e38f; i8[j] = 0; }
        float mx = 3.0e38f; int mp = 0;
        for (int g = 0; g < NWG * KNN; ++g) {
            float d = cd[g];
            if (d < mx) {
                int id = ci[g];
#pragma unroll
                for (int qq = 0; qq < KNN; ++qq)
                    if (qq == mp) { d8[qq] = d; i8[qq] = id; }
                mx = d8[0]; mp = 0;
#pragma unroll
                for (int qq = 1; qq < KNN; ++qq)
                    if (d8[qq] > mx) { mx = d8[qq]; mp = qq; }
            }
        }
        float s[KNN], sm = -3.0e38f;
#pragma unroll
        for (int j = 0; j < KNN; ++j) { s[j] = -d8[j] / TEMPK; sm = fmaxf(sm, s[j]); }
        float ssum = 0.f;
#pragma unroll
        for (int j = 0; j < KNN; ++j) { float e = expf(s[j] - sm); s[j] = e; ssum += e; }
#pragma unroll
        for (int j = 0; j < KNN; ++j) {
            stok_w[j] = s[j] / ssum;
            stok_id[j] = vals[i8[j]];
        }
    }
    __syncthreads();
    // softmax(base_logits) statistics
    const float* brow = base + (size_t)b * V_;
    float lm = -3.0e38f;
    for (int v = t; v < V_; v += 256) lm = fmaxf(lm, brow[v]);
    red[t] = lm; __syncthreads();
    for (int w = 128; w > 0; w >>= 1) { if (t < w) red[t] = fmaxf(red[t], red[t + w]); __syncthreads(); }
    const float bmaxv = red[0]; __syncthreads();
    float ls = 0.f;
    for (int v = t; v < V_; v += 256) ls += expf(brow[v] - bmaxv);
    red[t] = ls; __syncthreads();
    for (int w = 128; w > 0; w >>= 1) { if (t < w) red[t] += red[t + w]; __syncthreads(); }
    const float inv_bsum = 1.0f / red[0];

    float wj[KNN]; int tj[KNN];
#pragma unroll
    for (int j = 0; j < KNN; ++j) { wj[j] = stok_w[j]; tj[j] = stok_id[j]; }

    const float knorm = 1.0f / (1.0f + (float)V_ * EPSK);
    const size_t BV = (size_t)B_ * V_;
    float* o0 = out + (size_t)b * V_;
    float* o1 = o0 + BV;
    float* o2 = o1 + BV;
    for (int v = t; v < V_; v += 256) {
        float bl = brow[v];
        float pb = expf(bl - bmaxv) * inv_bsum;
        float praw = 0.f;
#pragma unroll
        for (int j = 0; j < KNN; ++j) praw += (tj[j] == v) ? wj[j] : 0.f;
        float pk = (praw + EPSK) * knorm;        // softmax(log(p+eps))
        o0[v] = logf((1.0f - LAM) * pb + LAM * pk);
        o1[v] = bl;
        o2[v] = logf(praw + EPSK);
    }
}

// ============ host-side launcher ============
extern "C" void kernel_launch(void* const* d_in, const int* in_sizes, int n_in,
                              void* d_out, int out_size, void* d_ws, size_t ws_size,
                              hipStream_t stream) {
    const float* q    = (const float*)d_in[0];   // [64,768]
    const float* base = (const float*)d_in[1];   // [64,50257]
    const float* keys = (const float*)d_in[2];   // [262144,768]
    const int*   vals = (const int*)d_in[3];     // [262144]
    float* out = (float*)d_out;                  // 3 x [64,50257]

    char* ws = (char*)d_ws;
    float*          cand_d = (float*)ws;                        // 256 KB
    int*            cand_i = (int*)(ws + 262144);               // 256 KB
    unsigned short* qbf    = (unsigned short*)(ws + 524288);    // 96 KB bf16 queries
    float*          q2g    = (float*)(ws + 524288 + 98304);     // 256 B

    convert_q   <<<B_,  256, 0, stream>>>(q, qbf, q2g);
    knn_gemm_topk<<<NWG, 256, 0, stream>>>(keys, qbf, q2g, cand_d, cand_i);
    knn_finalize<<<B_,  256, 0, stream>>>(base, vals, cand_d, cand_i, out);
}
